// MoE_66099546685736
// MI455X (gfx1250) — matrix operand order repaired and checked by reading, not compiled, not run on hardware
//
#include <hip/hip_runtime.h>
#include <hip/hip_bf16.h>
#include <stdint.h>

#define T_TOK 2048
#define NE    64
#define TOPK  8
#define DIM   512
#define IDIM  1024
#define CAP   1024

typedef __attribute__((ext_vector_type(16))) __bf16 bf16x16;
typedef __attribute__((ext_vector_type(8)))  float  floatx8;
typedef unsigned int uint32x4 __attribute__((ext_vector_type(4)));
typedef int          int32x4  __attribute__((ext_vector_type(4)));
typedef int          int32x8  __attribute__((ext_vector_type(8)));

union FragU { bf16x16 v; unsigned int d[8]; };

__device__ __forceinline__ unsigned short f2bf(float f) {
  unsigned int u = __float_as_uint(f);
  unsigned int r = u + 0x7FFFu + ((u >> 16) & 1u);   // round-to-nearest-even
  return (unsigned short)(r >> 16);
}

// --------------------------- Tensor Data Mover: 64x64 bf16 tile -> LDS -------
// D# per CDNA5 ISA. 2D tile, data_size=2B,
// tile_dim0=tile_dim1=64, tensor_dim0_stride = row stride (elements).
__device__ __forceinline__ void tdm_load_tile_bf16(const unsigned short* gtile,
                                                   unsigned int* lds_tile,
                                                   unsigned row_stride) {
  unsigned long long ga = (unsigned long long)(const void*)gtile;
  unsigned lds = (unsigned)(unsigned long long)(void*)lds_tile;  // low32 = LDS offset
  uint32x4 g0;
  g0.x = 1u;                                        // count=1, user descriptor
  g0.y = lds;                                       // lds_addr (bytes)
  g0.z = (unsigned)ga;                              // global_addr[31:0]
  g0.w = ((unsigned)(ga >> 32) & 0x01FFFFFFu) | (2u << 30);  // addr[56:32] | type=2
  const unsigned td1 = 1u << 20;                    // tensor_dim1: huge, no zero-fill
  int32x8 g1;
  g1.s0 = (int)(1u << 16);                          // wg_mask=0 | data_size=1 (2B)
  g1.s1 = (int)(row_stride << 16);                  // [63:48] tensor_dim0[15:0]
  g1.s2 = (int)((row_stride >> 16) | (td1 << 16));  // dim0[31:16] | dim1[15:0]
  g1.s3 = (int)((td1 >> 16) | (64u << 16));         // dim1[31:16] | tile_dim0=64
  g1.s4 = 64;                                       // tile_dim1=64 | tile_dim2=0
  g1.s5 = (int)row_stride;                          // tensor_dim0_stride[31:0]
  g1.s6 = 0;                                        // stride[47:32] | dim1_stride lo
  g1.s7 = 0;
  int32x4 z4 = {0, 0, 0, 0};
#if __clang_major__ >= 23
  int32x8 z8 = {0, 0, 0, 0, 0, 0, 0, 0};
  __builtin_amdgcn_tensor_load_to_lds(g0, g1, z4, z4, z8, 0);
#else
  __builtin_amdgcn_tensor_load_to_lds(g0, g1, z4, z4, 0);
#endif
}

// ---------------------------------------------------------------- gating ----
__global__ void zero_counts_kernel(int* counts) {
  if (threadIdx.x < NE) counts[threadIdx.x] = 0;
}

__global__ void gate_kernel(const float* __restrict__ x, const float* __restrict__ gw,
                            const float* __restrict__ bias,
                            int* __restrict__ counts, int* __restrict__ pair_e,
                            int* __restrict__ pair_slot,
                            int* __restrict__ slot2tok, float* __restrict__ slot2w) {
  __shared__ float xs[DIM];
  __shared__ float sc[NE];
  int t = blockIdx.x;
  int e = threadIdx.x;
  for (int j = e; j < DIM; j += 64) xs[j] = x[(size_t)t * DIM + j];
  __syncthreads();
  float s = bias[e];
  const float* gwe = gw + (size_t)e * DIM;
  for (int d = 0; d < DIM; ++d) s += xs[d] * gwe[d];
  sc[e] = s;
  __syncthreads();
  if (e == 0) {
    float mx = sc[0];
    for (int i = 1; i < NE; ++i) mx = fmaxf(mx, sc[i]);
    float Z = 0.f;
    for (int i = 0; i < NE; ++i) Z += __expf(sc[i] - mx);
    int sel[TOPK]; float pv[TOPK]; float psum = 0.f;
    for (int k = 0; k < TOPK; ++k) {
      int bi = 0; float bs = -1e30f;
      for (int i = 0; i < NE; ++i) if (sc[i] > bs) { bs = sc[i]; bi = i; }
      sel[k] = bi;
      pv[k] = __expf(bs - mx) / Z;
      psum += pv[k];
      sc[bi] = -1e30f;
    }
    float inv = 1.f / (psum + 1e-8f);
    for (int k = 0; k < TOPK; ++k) {
      int ee = sel[k];
      int slot = atomicAdd(&counts[ee], 1);
      int p = t * TOPK + k;
      pair_e[p] = ee;
      if (slot < CAP) {
        pair_slot[p] = slot;
        slot2tok[ee * CAP + slot] = t;
        slot2w[ee * CAP + slot] = pv[k] * inv;
      } else {
        pair_slot[p] = -1;
      }
    }
  }
}

// ------------------------------------------------------- dispatch / convert ----
__global__ void cvt_bf16_kernel(const float* __restrict__ src,
                                unsigned short* __restrict__ dst, int n) {
  int i = blockIdx.x * blockDim.x + threadIdx.x;
  if (i < n) dst[i] = f2bf(src[i]);
}

__global__ void gather_kernel(const float* __restrict__ x,
                              const int* __restrict__ pair_e,
                              const int* __restrict__ pair_slot,
                              unsigned short* __restrict__ Xb) {
  int p = blockIdx.x;
  int slot = pair_slot[p];
  if (slot < 0) return;
  int e = pair_e[p];
  int t = p >> 3;
  const float* src = x + (size_t)t * DIM;
  unsigned short* dst = Xb + ((size_t)e * CAP + slot) * DIM;
  for (int j = threadIdx.x; j < DIM; j += 128) dst[j] = f2bf(src[j]);
}

// --------------------------------------------------------- fragment helpers ----
__device__ __forceinline__ bf16x16 frag_A(const unsigned int* tile, int m0, int kc,
                                          int l15, int h16) {
  FragU f;
#pragma unroll
  for (int v = 0; v < 8; ++v) {
    int ka = kc + 2 * v + ((v & 4) ? 8 : 0) + h16 * 8;
    f.d[v] = tile[(m0 + l15) * 32 + (ka >> 1)];
  }
  return f.v;
}
__device__ __forceinline__ bf16x16 frag_B(const unsigned int* wt, int n0, int kc,
                                          int l15, int h16) {
  FragU f;
#pragma unroll
  for (int v = 0; v < 8; ++v) {
    int kb = kc + 2 * v + ((v & 4) ? 8 : 0) + h16 * 8;
    f.d[v] = wt[(n0 + l15) * 32 + (kb >> 1)];
  }
  return f.v;
}

// ------------------------------------------------------ SwiGLU up-projection ----
// H[m, n] = silu(X@W1)[m,n] * (X@W3)[m,n];  X bf16 [rows x DIM], W f32 [DIM x IDIM]
__global__ __launch_bounds__(256) void swiglu_kernel(
    const unsigned short* __restrict__ X, const float* __restrict__ W1,
    const float* __restrict__ W3, unsigned short* __restrict__ H,
    const int* __restrict__ counts, int mtilesPerE, int Mdense) {
  __shared__ unsigned int At[64 * 32];    // 64 rows x 64 bf16 (TDM destination)
  __shared__ unsigned int W1t[64 * 32];   // transposed [n][k]
  __shared__ unsigned int W3t[64 * 32];

  int e = 0, mt = blockIdx.x, rows = Mdense;
  size_t xoff = 0, woff = 0, hoff = 0;
  if (counts) {
    e = blockIdx.x / mtilesPerE;
    mt = blockIdx.x % mtilesPerE;
    rows = counts[e]; if (rows > CAP) rows = CAP;
    xoff = (size_t)e * CAP * DIM;
    woff = (size_t)e * DIM * IDIM;
    hoff = (size_t)e * CAP * IDIM;
  }
  int m_base = mt * 64;
  if (m_base >= rows) return;
  int n_base = blockIdx.y * 64;

  const unsigned short* Xp = X + xoff;
  const float* w1p = W1 + woff;
  const float* w3p = W3 + woff;
  unsigned short* Hp = H + hoff;

  int tid = threadIdx.x, lane = tid & 31, wave = tid >> 5;
  int wm = wave & 3, wn = wave >> 2;
  int h16 = lane >> 4, l15 = lane & 15;

  floatx8 acc1a = {}, acc1b = {}, acc3a = {}, acc3b = {};
  unsigned short* w1s = (unsigned short*)W1t;
  unsigned short* w3s = (unsigned short*)W3t;

  for (int kb = 0; kb < DIM; kb += 64) {
    // A tile via Tensor Data Mover (wave 0 issues; TDM ignores EXEC)
    if (wave == 0)
      tdm_load_tile_bf16(Xp + (size_t)m_base * DIM + kb, At, DIM);
    // W tiles: vectorized f32 loads, K-transposed bf16 into LDS
    {
      int n4 = (tid & 15) * 4;
      int kr = tid >> 4;                    // 0..15
#pragma unroll
      for (int j = 0; j < 4; ++j) {
        int k = j * 16 + kr;
        const float4 v1 = *(const float4*)&w1p[(size_t)(kb + k) * IDIM + n_base + n4];
        const float4 v3 = *(const float4*)&w3p[(size_t)(kb + k) * IDIM + n_base + n4];
        w1s[(n4 + 0) * 64 + k] = f2bf(v1.x);
        w1s[(n4 + 1) * 64 + k] = f2bf(v1.y);
        w1s[(n4 + 2) * 64 + k] = f2bf(v1.z);
        w1s[(n4 + 3) * 64 + k] = f2bf(v1.w);
        w3s[(n4 + 0) * 64 + k] = f2bf(v3.x);
        w3s[(n4 + 1) * 64 + k] = f2bf(v3.y);
        w3s[(n4 + 2) * 64 + k] = f2bf(v3.z);
        w3s[(n4 + 3) * 64 + k] = f2bf(v3.w);
      }
    }
    if (kb + 64 < DIM) {
      __builtin_prefetch(&w1p[(size_t)(kb + 64) * IDIM + n_base], 0, 1);
      __builtin_prefetch(&w3p[(size_t)(kb + 64) * IDIM + n_base], 0, 1);
    }
    if (wave == 0) __builtin_amdgcn_s_wait_tensorcnt(0);
    __syncthreads();
#pragma unroll
    for (int kc = 0; kc < 64; kc += 32) {
      bf16x16 a  = frag_A(At, wm * 16, kc, l15, h16);
      bf16x16 b0 = frag_B(W1t, wn * 32,      kc, l15, h16);
      bf16x16 b1 = frag_B(W1t, wn * 32 + 16, kc, l15, h16);
      bf16x16 c0 = frag_B(W3t, wn * 32,      kc, l15, h16);
      bf16x16 c1 = frag_B(W3t, wn * 32 + 16, kc, l15, h16);
      acc1a = __builtin_amdgcn_wmma_f32_16x16x32_bf16(false, a, false, b0, (short)0, acc1a, false, false);
      acc1b = __builtin_amdgcn_wmma_f32_16x16x32_bf16(false, a, false, b1, (short)0, acc1b, false, false);
      acc3a = __builtin_amdgcn_wmma_f32_16x16x32_bf16(false, a, false, c0, (short)0, acc3a, false, false);
      acc3b = __builtin_amdgcn_wmma_f32_16x16x32_bf16(false, a, false, c1, (short)0, acc3b, false, false);
    }
    __syncthreads();
  }
#pragma unroll
  for (int v = 0; v < 8; ++v) {
    int row = m_base + wm * 16 + v + 8 * h16;
    if (row < rows) {
      int c0 = n_base + wn * 32 + l15;
      float g1 = acc1a[v], g3 = acc3a[v];
      Hp[(size_t)row * IDIM + c0] = f2bf((g1 / (1.f + __expf(-g1))) * g3);
      float q1 = acc1b[v], q3 = acc3b[v];
      Hp[(size_t)row * IDIM + c0 + 16] = f2bf((q1 / (1.f + __expf(-q1))) * q3);
    }
  }
}

// ---------------------------------------------- down-projection (+ combine) ----
// Y = H @ W2 ; expert mode: atomicAdd(out[tok], w * Y[slot]) ; dense: out = Y
__global__ __launch_bounds__(256) void down_kernel(
    const unsigned short* __restrict__ H, const float* __restrict__ W2,
    float* __restrict__ out, const int* __restrict__ counts,
    const int* __restrict__ slot2tok, const float* __restrict__ slot2w,
    int mtilesPerE, int Mdense) {
  __shared__ unsigned int At[64 * 32];   // TDM destination
  __shared__ unsigned int Wt[64 * 32];

  int e = 0, mt = blockIdx.x, rows = Mdense;
  size_t hoff = 0, woff = 0;
  if (counts) {
    e = blockIdx.x / mtilesPerE;
    mt = blockIdx.x % mtilesPerE;
    rows = counts[e]; if (rows > CAP) rows = CAP;
    hoff = (size_t)e * CAP * IDIM;
    woff = (size_t)e * IDIM * DIM;
  }
  int m_base = mt * 64;
  if (m_base >= rows) return;
  int n_base = blockIdx.y * 64;

  const unsigned short* Hbase = H + hoff;
  const float* w2p = W2 + woff;

  int tid = threadIdx.x, lane = tid & 31, wave = tid >> 5;
  int wm = wave & 3, wn = wave >> 2;
  int h16 = lane >> 4, l15 = lane & 15;

  floatx8 acca = {}, accb = {};
  unsigned short* wts = (unsigned short*)Wt;

  for (int kb = 0; kb < IDIM; kb += 64) {
    if (wave == 0)
      tdm_load_tile_bf16(Hbase + (size_t)m_base * IDIM + kb, At, IDIM);
    {
      int n4 = (tid & 15) * 4;
      int kr = tid >> 4;
#pragma unroll
      for (int j = 0; j < 4; ++j) {
        int k = j * 16 + kr;
        const float4 v2 = *(const float4*)&w2p[(size_t)(kb + k) * DIM + n_base + n4];
        wts[(n4 + 0) * 64 + k] = f2bf(v2.x);
        wts[(n4 + 1) * 64 + k] = f2bf(v2.y);
        wts[(n4 + 2) * 64 + k] = f2bf(v2.z);
        wts[(n4 + 3) * 64 + k] = f2bf(v2.w);
      }
    }
    if (kb + 64 < IDIM)
      __builtin_prefetch(&w2p[(size_t)(kb + 64) * DIM + n_base], 0, 1);
    if (wave == 0) __builtin_amdgcn_s_wait_tensorcnt(0);
    __syncthreads();
#pragma unroll
    for (int kc = 0; kc < 64; kc += 32) {
      bf16x16 a  = frag_A(At, wm * 16, kc, l15, h16);
      bf16x16 b0 = frag_B(Wt, wn * 32,      kc, l15, h16);
      bf16x16 b1 = frag_B(Wt, wn * 32 + 16, kc, l15, h16);
      acca = __builtin_amdgcn_wmma_f32_16x16x32_bf16(false, a, false, b0, (short)0, acca, false, false);
      accb = __builtin_amdgcn_wmma_f32_16x16x32_bf16(false, a, false, b1, (short)0, accb, false, false);
    }
    __syncthreads();
  }
#pragma unroll
  for (int v = 0; v < 8; ++v) {
    int row = m_base + wm * 16 + v + 8 * h16;
    if (row < rows) {
      int c0 = n_base + wn * 32 + l15;
      float y0 = acca[v], y1 = accb[v];
      if (counts) {
        int tok = slot2tok[e * CAP + row];
        float w = slot2w[e * CAP + row];
        atomicAdd(&out[(size_t)tok * DIM + c0], y0 * w);
        atomicAdd(&out[(size_t)tok * DIM + c0 + 16], y1 * w);
      } else {
        out[(size_t)row * DIM + c0] = y0;
        out[(size_t)row * DIM + c0 + 16] = y1;
      }
    }
  }
}

// ------------------------------------------------------------------ launch ----
extern "C" void kernel_launch(void* const* d_in, const int* in_sizes, int n_in,
                              void* d_out, int out_size, void* d_ws, size_t ws_size,
                              hipStream_t stream) {
  (void)in_sizes; (void)n_in; (void)out_size; (void)ws_size;
  const float* x    = (const float*)d_in[0];
  const float* gw   = (const float*)d_in[1];
  const float* bias = (const float*)d_in[2];
  const float* w1   = (const float*)d_in[3];
  const float* w3   = (const float*)d_in[4];
  const float* w2   = (const float*)d_in[5];
  const float* ws1  = (const float*)d_in[6];
  const float* ws3  = (const float*)d_in[7];
  const float* ws2  = (const float*)d_in[8];
  float* out = (float*)d_out;

  char* base = (char*)d_ws;
  size_t o = 0;
  auto alloc = [&](size_t bytes) -> char* {
    char* p = base + o;
    o += (bytes + 255) & ~(size_t)255;
    return p;
  };
  int*            counts    = (int*)alloc((size_t)NE * 4);
  int*            pair_e    = (int*)alloc((size_t)T_TOK * TOPK * 4);
  int*            pair_slot = (int*)alloc((size_t)T_TOK * TOPK * 4);
  int*            slot2tok  = (int*)alloc((size_t)NE * CAP * 4);
  float*          slot2w    = (float*)alloc((size_t)NE * CAP * 4);
  unsigned short* Xb        = (unsigned short*)alloc((size_t)NE * CAP * DIM * 2);
  unsigned short* Xs        = (unsigned short*)alloc((size_t)T_TOK * DIM * 2);
  unsigned short* He        = (unsigned short*)alloc((size_t)NE * CAP * IDIM * 2);
  unsigned short* Hs        = (unsigned short*)alloc((size_t)T_TOK * IDIM * 2);

  hipLaunchKernelGGL(zero_counts_kernel, dim3(1), dim3(64), 0, stream, counts);
  hipLaunchKernelGGL(gate_kernel, dim3(T_TOK), dim3(64), 0, stream,
                     x, gw, bias, counts, pair_e, pair_slot, slot2tok, slot2w);
  hipLaunchKernelGGL(cvt_bf16_kernel, dim3((T_TOK * DIM + 255) / 256), dim3(256), 0, stream,
                     x, Xs, T_TOK * DIM);
  hipLaunchKernelGGL(gather_kernel, dim3(T_TOK * TOPK), dim3(128), 0, stream,
                     x, pair_e, pair_slot, Xb);
  // shared expert (down_kernel dense mode initializes `out`)
  hipLaunchKernelGGL(swiglu_kernel, dim3(T_TOK / 64, IDIM / 64), dim3(256), 0, stream,
                     Xs, ws1, ws3, Hs, (const int*)nullptr, 1, T_TOK);
  hipLaunchKernelGGL(down_kernel, dim3(T_TOK / 64, DIM / 64), dim3(256), 0, stream,
                     Hs, ws2, out, (const int*)nullptr, (const int*)nullptr,
                     (const float*)nullptr, 1, T_TOK);
  // routed experts (atomic weighted combine on top of shared output)
  hipLaunchKernelGGL(swiglu_kernel, dim3(NE * (CAP / 64), IDIM / 64), dim3(256), 0, stream,
                     Xb, w1, w3, He, counts, CAP / 64, CAP);
  hipLaunchKernelGGL(down_kernel, dim3(NE * (CAP / 64), DIM / 64), dim3(256), 0, stream,
                     He, w2, out, counts, slot2tok, slot2w, CAP / 64, CAP);
}
